// RelativeMultiHeadAttention_31945966747947
// MI455X (gfx1250) — compile-verified, matrix operand order checked by reading, not run on hardware
//
#include <hip/hip_runtime.h>
#include <stdint.h>

#define DM    1024
#define NH    16
#define DH    64
#define LSEQ  2048
#define BSZ   2
#define NEMB  1025
#define MAXREL 512

typedef __attribute__((ext_vector_type(16))) __bf16 v16bf;
typedef __attribute__((ext_vector_type(8)))  float  v8f;
typedef __attribute__((ext_vector_type(4)))  unsigned int v4u;
typedef __attribute__((ext_vector_type(8)))  int v8i;
typedef __attribute__((ext_vector_type(4)))  int v4i;

#define V8F_ZERO {0.f,0.f,0.f,0.f,0.f,0.f,0.f,0.f}

union Frag16 { v16bf v; unsigned short h[16]; };

__device__ __forceinline__ unsigned short f2bf(float x) {
  uint32_t u = __builtin_bit_cast(uint32_t, x);
  u += 0x7fffu + ((u >> 16) & 1u);           // round-to-nearest-even
  return (unsigned short)(u >> 16);
}
__device__ __forceinline__ float bf2f(unsigned short h) {
  uint32_t u = ((uint32_t)h) << 16;
  return __builtin_bit_cast(float, u);
}

__device__ __forceinline__ v8f wmma_bf16(v16bf a, v16bf b, v8f c) {
  // D = A(16x32) * B(32x16) + C, f32 accumulate
  return __builtin_amdgcn_wmma_f32_16x16x32_bf16(false, a, false, b, (short)0, c, false, false);
}

// Load one 16x32 bf16 operand fragment (A or B role) from a row-major ushort
// buffer. 'row' must already include the (lane&15) term. Per ISA 7.12.2:
// lanes 0-15 hold K {k0..k0+7, k0+16..k0+23}, lanes 16-31 hold K {+8, +24}.
__device__ __forceinline__ v16bf load_frag(const unsigned short* p, int row, int ld,
                                           int k0, int lane) {
  const unsigned short* s = p + (size_t)row * ld + k0 + ((lane >> 4) << 3);
  Frag16 f;
#pragma unroll
  for (int e = 0; e < 8; ++e) { f.h[e] = s[e]; f.h[8 + e] = s[16 + e]; }
  return f.v;
}

// ---------------------------------------------------------------------------
// TDM: issue one 2D tile load (Tensor Data Mover) Global -> LDS.
// Descriptor packing per CDNA5 ISA 8.3/8.4 (D# group0 = 128b, group1 = 256b).
//   tile_d0 elements per row (2B each), tile_d1 rows, row stride in elements.
// Whole-wave operation (EXEC ignored); completion tracked with TENSORcnt.
// This toolchain exposes the 6-arg builtin (extra int32x8 descriptor group).
// ---------------------------------------------------------------------------
__device__ __forceinline__ void tdm_load_2d(unsigned lds_off, const void* gptr,
                                            unsigned tile_d0, unsigned tile_d1,
                                            unsigned long long row_stride) {
  unsigned long long ga = (unsigned long long)(uintptr_t)gptr;
  v4u g0;
  g0[0] = 1u;                                        // count=1, user mode
  g0[1] = lds_off;                                   // LDS byte address
  g0[2] = (unsigned)ga;                              // global_addr[31:0]
  g0[3] = (unsigned)((ga >> 32) & 0x01ffffffu) | (2u << 30);  // [56:32], type=2
  v8i g1;
  g1[0] = 0x00010000;                                // data_size=1 (2B), mask=0
  g1[1] = (int)((tile_d0 & 0xffffu) << 16);          // tensor_dim0 = tile_d0
  g1[2] = (int)(((tile_d1 & 0xffffu) << 16));        // tensor_dim1 = tile_d1
  g1[3] = (int)(tile_d0 << 16);                      // tile_dim0
  g1[4] = (int)(tile_d1 & 0xffffu);                  // tile_dim1 (tile_dim2=0)
  g1[5] = (int)(unsigned)row_stride;                 // tensor_dim0_stride[31:0]
  g1[6] = (int)(unsigned)(row_stride >> 32);         // stride[47:32]
  g1[7] = 0;
  v4i z4 = {0, 0, 0, 0};
  v8i z8 = {0, 0, 0, 0, 0, 0, 0, 0};
  __builtin_amdgcn_tensor_load_to_lds(g0, g1, z4, z4, z8, 0);
}

// ---------------------------------------------------------------------------
// Kernel 1: convert weights to transposed bf16 (Wt[n][k]) and rel_emb to bf16
// ---------------------------------------------------------------------------
__global__ __launch_bounds__(256) void convert_params(
    const float* __restrict__ Wq, const float* __restrict__ Wk,
    const float* __restrict__ Wv, const float* __restrict__ Wo,
    const float* __restrict__ rel,
    unsigned short* __restrict__ wt, unsigned short* __restrict__ relw) {
  int idx = blockIdx.x * 256 + threadIdx.x;
  const int NW = 4 * DM * DM;
  if (idx < NW) {
    int which = idx >> 20;                 // DM*DM == 1<<20
    int rem   = idx & (DM * DM - 1);
    int k = rem >> 10, n = rem & 1023;
    const float* W = (which == 0) ? Wq : (which == 1) ? Wk : (which == 2) ? Wv : Wo;
    wt[(size_t)which * DM * DM + (size_t)n * DM + k] = f2bf(W[rem]);
  } else {
    int r = idx - NW;
    if (r < NEMB * DH) relw[r] = f2bf(rel[r]);
  }
}

// ---------------------------------------------------------------------------
// Kernel 2: QKV projections. z = 0/1/2 -> Q/K/V.
// Q,K -> [B,H,L,64] bf16.  V -> transposed [B,H,64,L] bf16.
// ---------------------------------------------------------------------------
__global__ __launch_bounds__(256) void proj_gemm(
    const float* __restrict__ q_in, const float* __restrict__ k_in,
    const float* __restrict__ v_in, const unsigned short* __restrict__ wt,
    const float* __restrict__ bq, const float* __restrict__ bk,
    const float* __restrict__ bv,
    unsigned short* __restrict__ Qbf, unsigned short* __restrict__ Kbf,
    unsigned short* __restrict__ Vt) {
  const int z = blockIdx.z;
  const float* X = (z == 0) ? q_in : (z == 1) ? k_in : v_in;
  const unsigned short* W = wt + (size_t)z * DM * DM;   // transposed [n][k]
  const float* bias = (z == 0) ? bq : (z == 1) ? bk : bv;
  const int m0 = blockIdx.x * 64;
  const int n0 = blockIdx.y * 64;
  const int tid = threadIdx.x, lane = tid & 31, wave = tid >> 5;

  __shared__ __align__(16) unsigned short sA[64 * 32];
  __shared__ __align__(16) unsigned short sB[64 * 32];

  const int nsub  = wave & 3;
  const int msub0 = (wave >> 2) * 2;
  v8f acc0 = V8F_ZERO, acc1 = V8F_ZERO;

  for (int k0 = 0; k0 < DM; k0 += 32) {
    __syncthreads();
    { // stage A tile 64x32, f32 -> bf16
      int i0 = tid * 8;
      int row = i0 >> 5, col = i0 & 31;
      const float* src = X + (size_t)(m0 + row) * DM + k0 + col;
#pragma unroll
      for (int e = 0; e < 8; ++e) sA[row * 32 + col + e] = f2bf(src[e]);
      if (k0 + 32 < DM) __builtin_prefetch(src + 32, 0, 1);  // next K-step
    }
    { // stage B tile 64x32 bf16 (already transposed: rows = n, contiguous k)
      int row = tid >> 2, c = (tid & 3) * 8;
      const unsigned short* src = W + (size_t)(n0 + row) * DM + k0 + c;
      *(uint4*)(&sB[row * 32 + c]) = *(const uint4*)src;
      if (k0 + 32 < DM) __builtin_prefetch(src + 32, 0, 1);
    }
    __syncthreads();
    v16bf bfr = load_frag(sB, nsub * 16 + (lane & 15), 32, 0, lane);
    v16bf a0  = load_frag(sA, (msub0 + 0) * 16 + (lane & 15), 32, 0, lane);
    v16bf a1  = load_frag(sA, (msub0 + 1) * 16 + (lane & 15), 32, 0, lane);
    acc0 = wmma_bf16(a0, bfr, acc0);
    acc1 = wmma_bf16(a1, bfr, acc1);
  }

  const int n  = n0 + nsub * 16 + (lane & 15);
  const float bb = bias[n];
  const int h = n >> 6, dh = n & 63;
  const int mrow = (lane >> 4) << 3;
#pragma unroll
  for (int e = 0; e < 8; ++e) {
    int gm0 = m0 + msub0 * 16 + e + mrow;
    int gm1 = gm0 + 16;
    float v0 = acc0[e] + bb, v1 = acc1[e] + bb;
    int b0_ = gm0 >> 11, l0 = gm0 & 2047;
    int b1_ = gm1 >> 11, l1 = gm1 & 2047;
    if (z == 0) {
      Qbf[(((size_t)(b0_ * NH + h)) * LSEQ + l0) * DH + dh] = f2bf(v0);
      Qbf[(((size_t)(b1_ * NH + h)) * LSEQ + l1) * DH + dh] = f2bf(v1);
    } else if (z == 1) {
      Kbf[(((size_t)(b0_ * NH + h)) * LSEQ + l0) * DH + dh] = f2bf(v0);
      Kbf[(((size_t)(b1_ * NH + h)) * LSEQ + l1) * DH + dh] = f2bf(v1);
    } else {
      Vt[(((size_t)(b0_ * NH + h)) * DH + dh) * LSEQ + l0] = f2bf(v0);
      Vt[(((size_t)(b1_ * NH + h)) * DH + dh) * LSEQ + l1] = f2bf(v1);
    }
  }
}

// ---------------------------------------------------------------------------
// Kernel 3: flash-style attention with relative-position window.
// Block = 128 query rows (8 waves x 16), loops over 32 key tiles of 64.
// K/V tiles are staged in LDS by the Tensor Data Mover (wave 0 issues the
// DMA, all waves compute the position-window WMMAs while it flies, then the
// issuing wave drains TENSORcnt before the workgroup barrier).
// ---------------------------------------------------------------------------
__global__ __launch_bounds__(256) void attn_kernel(
    const unsigned short* __restrict__ Qbf, const unsigned short* __restrict__ Kbf,
    const unsigned short* __restrict__ Vt,  const unsigned short* __restrict__ relw,
    unsigned short* __restrict__ abuf) {
  const int qt = blockIdx.x, h = blockIdx.y, b = blockIdx.z;
  const int tid = threadIdx.x, lane = tid & 31, wave = tid >> 5;
  const int iw = qt * 128 + wave * 16;                 // this wave's 16 query rows
  const unsigned short* Qb = Qbf + ((size_t)(b * NH + h)) * LSEQ * DH;
  const unsigned short* Kb = Kbf + ((size_t)(b * NH + h)) * LSEQ * DH;
  const unsigned short* Vb = Vt  + ((size_t)(b * NH + h)) * DH * LSEQ;

  __shared__ __align__(16) unsigned short sK[64 * 64];       // K tile [j][d]
  __shared__ __align__(16) unsigned short sV[64 * 64];       // V tile [d][j]
  __shared__ __align__(16) unsigned short sP[8][16 * 64];    // per-wave P bf16
  __shared__ __align__(16) unsigned short sPos[8][16 * 96];  // per-wave pos window

  // LDS byte offsets for the TDM descriptors (generic LDS addr low 32 bits)
  const unsigned sK_off = (unsigned)(uintptr_t)(&sK[0]);
  const unsigned sV_off = (unsigned)(uintptr_t)(&sV[0]);

  // Loop-invariant Q A-operand fragments (16x64 over two K=32 chunks)
  const v16bf aQ0 = load_frag(Qb, iw + (lane & 15), DH, 0, lane);
  const v16bf aQ1 = load_frag(Qb, iw + (lane & 15), DH, 32, lane);

  v8f O[4]; float rmax[8], rsum[8];
#pragma unroll
  for (int d = 0; d < 4; ++d) O[d] = (v8f)V8F_ZERO;
#pragma unroll
  for (int e = 0; e < 8; ++e) { rmax[e] = -3.0e38f; rsum[e] = 0.f; }

  const float SCALE = 0.125f;            // 1/sqrt(64)
  const int mrow = (lane >> 4) << 3;     // +e gives C/D row M

  for (int jt = 0; jt < LSEQ / 64; ++jt) {
    const int j0 = jt * 64;
    __syncthreads();                     // previous iter done with sK/sV
    if (wave == 0) {
      // K tile: 64 rows x 64 bf16, row stride 64 elements
      tdm_load_2d(sK_off, Kb + (size_t)j0 * DH, 64, 64, 64ull);
      // V tile (transposed source): 64 dh-rows x 64 bf16, row stride LSEQ
      tdm_load_2d(sV_off, Vb + j0, 64, 64, (unsigned long long)LSEQ);
    }

    // Relative-position window: pos(m,c) = q_{iw+m} . rel[clamp(rbase+c)]
    // (global + sPos only -> overlaps with the TDM fill of sK/sV)
    const int rbase = j0 - iw - 15 + MAXREL;
#pragma unroll
    for (int cs = 0; cs < 6; ++cs) {
      int r = rbase + cs * 16 + (lane & 15);
      r = (r < 0) ? 0 : ((r > 2 * MAXREL) ? 2 * MAXREL : r);
      v16bf b0 = load_frag(relw, r, DH, 0, lane);
      v16bf b1 = load_frag(relw, r, DH, 32, lane);
      v8f pc = V8F_ZERO;
      pc = wmma_bf16(aQ0, b0, pc);
      pc = wmma_bf16(aQ1, b1, pc);
#pragma unroll
      for (int e = 0; e < 8; ++e)
        sPos[wave][(e + mrow) * 96 + cs * 16 + (lane & 15)] = f2bf(pc[e]);
    }
    if (wave == 0) __builtin_amdgcn_s_wait_tensorcnt(0);  // DMA landed in LDS
    __syncthreads();                     // sK/sV + sPos visible to all waves

    // Content scores + gathered position scores
    v8f S[4];
#pragma unroll
    for (int ns = 0; ns < 4; ++ns) {
      v16bf bk0 = load_frag(sK, ns * 16 + (lane & 15), DH, 0, lane);
      v16bf bk1 = load_frag(sK, ns * 16 + (lane & 15), DH, 32, lane);
      v8f s = V8F_ZERO;
      s = wmma_bf16(aQ0, bk0, s);
      s = wmma_bf16(aQ1, bk1, s);
#pragma unroll
      for (int e = 0; e < 8; ++e) {
        int m = e + mrow;
        int c = ns * 16 + (lane & 15) + 15 - m;       // in [0,78]
        S[ns][e] = (s[e] + bf2f(sPos[wave][m * 96 + c])) * SCALE;
      }
    }

    // Online softmax (row = 16 lanes of a half-wave)
    float alpha[8];
#pragma unroll
    for (int e = 0; e < 8; ++e) {
      float mx = fmaxf(fmaxf(S[0][e], S[1][e]), fmaxf(S[2][e], S[3][e]));
#pragma unroll
      for (int off = 1; off < 16; off <<= 1) mx = fmaxf(mx, __shfl_xor(mx, off, 32));
      float mnew = fmaxf(rmax[e], mx);
      alpha[e] = __expf(rmax[e] - mnew);
      rmax[e] = mnew;
    }
#pragma unroll
    for (int ns = 0; ns < 4; ++ns)
#pragma unroll
      for (int e = 0; e < 8; ++e) S[ns][e] = __expf(S[ns][e] - rmax[e]);
#pragma unroll
    for (int e = 0; e < 8; ++e) {
      float sm = S[0][e] + S[1][e] + S[2][e] + S[3][e];
#pragma unroll
      for (int off = 1; off < 16; off <<= 1) sm += __shfl_xor(sm, off, 32);
      rsum[e] = rsum[e] * alpha[e] + sm;
    }
#pragma unroll
    for (int ds = 0; ds < 4; ++ds)
#pragma unroll
      for (int e = 0; e < 8; ++e) O[ds][e] *= alpha[e];

    // P -> LDS (bf16) so it can be re-read in A-operand layout
#pragma unroll
    for (int ns = 0; ns < 4; ++ns)
#pragma unroll
      for (int e = 0; e < 8; ++e)
        sP[wave][(e + mrow) * 64 + ns * 16 + (lane & 15)] = f2bf(S[ns][e]);
    __syncthreads();

    // O += P(16x64) @ V(64x64)
    v16bf aP0 = load_frag(sP[wave], (lane & 15), 64, 0, lane);
    v16bf aP1 = load_frag(sP[wave], (lane & 15), 64, 32, lane);
#pragma unroll
    for (int ds = 0; ds < 4; ++ds) {
      v16bf bv0 = load_frag(sV, ds * 16 + (lane & 15), 64, 0, lane);
      v16bf bv1 = load_frag(sV, ds * 16 + (lane & 15), 64, 32, lane);
      O[ds] = wmma_bf16(aP0, bv0, O[ds]);
      O[ds] = wmma_bf16(aP1, bv1, O[ds]);
    }
  }

  // Normalize and write merged-head output [B][L][H*64] as bf16
#pragma unroll
  for (int ds = 0; ds < 4; ++ds)
#pragma unroll
    for (int e = 0; e < 8; ++e) {
      int m = e + mrow;
      float val = O[ds][e] / rsum[e];
      int col = h * 64 + ds * 16 + (lane & 15);
      abuf[((size_t)b * LSEQ + (iw + m)) * DM + col] = f2bf(val);
    }
}

// ---------------------------------------------------------------------------
// Kernel 4: output projection, bf16 A x bf16 Wo^T + bo -> f32 d_out
// ---------------------------------------------------------------------------
__global__ __launch_bounds__(256) void out_gemm(
    const unsigned short* __restrict__ A, const unsigned short* __restrict__ Wt,
    const float* __restrict__ bias, float* __restrict__ out) {
  const int m0 = blockIdx.x * 64, n0 = blockIdx.y * 64;
  const int tid = threadIdx.x, lane = tid & 31, wave = tid >> 5;
  __shared__ __align__(16) unsigned short sA[64 * 32];
  __shared__ __align__(16) unsigned short sB[64 * 32];
  const int nsub = wave & 3, msub0 = (wave >> 2) * 2;
  v8f acc0 = V8F_ZERO, acc1 = V8F_ZERO;

  for (int k0 = 0; k0 < DM; k0 += 32) {
    __syncthreads();
    {
      int row = tid >> 2, c = (tid & 3) * 8;
      const unsigned short* srcA = A  + (size_t)(m0 + row) * DM + k0 + c;
      const unsigned short* srcB = Wt + (size_t)(n0 + row) * DM + k0 + c;
      *(uint4*)(&sA[row * 32 + c]) = *(const uint4*)srcA;
      *(uint4*)(&sB[row * 32 + c]) = *(const uint4*)srcB;
      if (k0 + 32 < DM) {
        __builtin_prefetch(srcA + 32, 0, 1);
        __builtin_prefetch(srcB + 32, 0, 1);
      }
    }
    __syncthreads();
    v16bf bfr = load_frag(sB, nsub * 16 + (lane & 15), 32, 0, lane);
    v16bf a0  = load_frag(sA, (msub0 + 0) * 16 + (lane & 15), 32, 0, lane);
    v16bf a1  = load_frag(sA, (msub0 + 1) * 16 + (lane & 15), 32, 0, lane);
    acc0 = wmma_bf16(a0, bfr, acc0);
    acc1 = wmma_bf16(a1, bfr, acc1);
  }

  const int n = n0 + nsub * 16 + (lane & 15);
  const float bb = bias[n];
  const int mrow = (lane >> 4) << 3;
#pragma unroll
  for (int e = 0; e < 8; ++e) {
    int m = m0 + msub0 * 16 + e + mrow;
    out[(size_t)m * DM + n]        = acc0[e] + bb;
    out[(size_t)(m + 16) * DM + n] = acc1[e] + bb;
  }
}

// ---------------------------------------------------------------------------
extern "C" void kernel_launch(void* const* d_in, const int* in_sizes, int n_in,
                              void* d_out, int out_size, void* d_ws, size_t ws_size,
                              hipStream_t stream) {
  const float* q   = (const float*)d_in[0];
  const float* k   = (const float*)d_in[1];
  const float* v   = (const float*)d_in[2];
  const float* Wq  = (const float*)d_in[3];
  const float* bq  = (const float*)d_in[4];
  const float* Wk  = (const float*)d_in[5];
  const float* bk  = (const float*)d_in[6];
  const float* Wv  = (const float*)d_in[7];
  const float* bv  = (const float*)d_in[8];
  const float* Wo  = (const float*)d_in[9];
  const float* bo  = (const float*)d_in[10];
  const float* rel = (const float*)d_in[11];
  float* out = (float*)d_out;

  // Workspace layout (bf16 stored as ushort), total ~40.3 MB:
  char* ws = (char*)d_ws;
  unsigned short* wt   = (unsigned short*)ws;                           // 4 x 2 MB
  unsigned short* relw = (unsigned short*)(ws + (8ull << 20));          // 128 KB
  unsigned short* Qbf  = (unsigned short*)(ws + (8ull << 20) + (1ull << 18));
  const size_t PER = (size_t)BSZ * NH * LSEQ * DH;                      // 4M halfs
  unsigned short* Kbf  = Qbf + PER;
  unsigned short* Vt   = Kbf + PER;
  unsigned short* abuf = Vt + PER;                                      // 8 MB

  const int convN = 4 * DM * DM + NEMB * DH;
  convert_params<<<(convN + 255) / 256, 256, 0, stream>>>(Wq, Wk, Wv, Wo, rel, wt, relw);
  proj_gemm<<<dim3(4096 / 64, DM / 64, 3), 256, 0, stream>>>(q, k, v, wt, bq, bk, bv,
                                                             Qbf, Kbf, Vt);
  attn_kernel<<<dim3(LSEQ / 128, NH, BSZ), 256, 0, stream>>>(Qbf, Kbf, Vt, relw, abuf);
  out_gemm<<<dim3(4096 / 64, DM / 64), 256, 0, stream>>>(abuf, wt + 3ull * DM * DM, bo, out);
}
